// DGI_GIN_10273561772527
// MI455X (gfx1250) — compile-verified, matrix operand order checked by reading, not run on hardware
//
#include <hip/hip_runtime.h>
#include <hip/hip_bf16.h>

typedef __attribute__((ext_vector_type(16))) _Float16 v16h;
typedef __attribute__((ext_vector_type(8)))  _Float16 v8h;
typedef __attribute__((ext_vector_type(8)))  float    v8f;

union V16U { v16h v; v8h h[2]; };

// ---------------------------------------------------------------- utilities

__global__ void zero_b128_kernel(float4* __restrict__ p, long long n4) {
    long long i = (long long)blockIdx.x * blockDim.x + threadIdx.x;
    if (i < n4) p[i] = make_float4(0.f, 0.f, 0.f, 0.f);
}

__global__ void deg_int_kernel(const int* __restrict__ dst, int* __restrict__ deg, int E) {
    int i = blockIdx.x * blockDim.x + threadIdx.x;
    if (i < E) atomicAdd(&deg[dst[i]], 1);
}

__global__ void invdeg_kernel(const int* __restrict__ deg, float* __restrict__ invdeg, int N) {
    int i = blockIdx.x * blockDim.x + threadIdx.x;
    if (i < N) invdeg[i] = 1.0f / (float)max(deg[i], 1);
}

__global__ void f32_to_f16_kernel(const float* __restrict__ in, _Float16* __restrict__ out, int n) {
    int i = blockIdx.x * blockDim.x + threadIdx.x;
    if (i < n) out[i] = (_Float16)in[i];
}

__global__ void copy_i32_kernel(const int* __restrict__ in, int* __restrict__ out, int n) {
    int i = blockIdx.x * blockDim.x + threadIdx.x;
    if (i < n) out[i] = in[i];
}

// ------------------------------------------------ exclusive scan (CSR row_ptr)
// Single block, chunked Hillis-Steele. N=100K -> 98 chunks of 1024; negligible
// cost vs the edge phases, trivially graph-capture safe.

__global__ void __launch_bounds__(1024)
exscan_kernel(const int* __restrict__ deg, int* __restrict__ rowptr, int N) {
    __shared__ int buf[1024];
    __shared__ int carry_s;
    const int t = threadIdx.x;
    if (t == 0) carry_s = 0;
    __syncthreads();
    for (int base = 0; base < N; base += 1024) {
        int i = base + t;
        int x = (i < N) ? deg[i] : 0;
        buf[t] = x;
        __syncthreads();
        for (int off = 1; off < 1024; off <<= 1) {
            int v = (t >= off) ? buf[t - off] : 0;
            __syncthreads();
            buf[t] += v;
            __syncthreads();
        }
        int incl = buf[t];
        int c = carry_s;                    // everyone reads old carry
        if (i < N) rowptr[i] = c + incl - x; // exclusive
        __syncthreads();
        if (t == 0) carry_s = c + buf[1023];
        __syncthreads();
    }
    if (t == 0) rowptr[N] = carry_s;
}

__global__ void csr_fill_kernel(const int* __restrict__ src, const int* __restrict__ dst,
                                int* __restrict__ cursor, int* __restrict__ csrsrc, int E) {
    int i = blockIdx.x * blockDim.x + threadIdx.x;
    if (i < E) {
        int pos = atomicAdd(&cursor[dst[i]], 1);
        csrsrc[pos] = src[i];
    }
}

// ------------------------------------------------ pull-mode aggregation (no float atomics)
// thread = (dst node, 4-feature group); loop over in-neighbors, float4 gathers.
// Lanes of a wave cover consecutive features -> each neighbor row is a
// coalesced 512B-per-wave read, L2-resident (h <= 102MB < 192MB L2). agg written once.

__global__ void aggregate_csr_kernel(const float* __restrict__ hin,
                                     const int* __restrict__ rowptr,
                                     const int* __restrict__ csrsrc,
                                     float* __restrict__ agg,
                                     long long total, int F4, int F) {
    long long idx = (long long)blockIdx.x * blockDim.x + threadIdx.x;
    if (idx >= total) return;
    int n = (int)(idx / F4);
    int f = (int)(idx % F4) * 4;
    int beg = rowptr[n], end = rowptr[n + 1];
    float4 acc = make_float4(0.f, 0.f, 0.f, 0.f);
    for (int j = beg; j < end; ++j) {
        int s = csrsrc[j];                                  // broadcast within node's lanes
        const float4 v = *(const float4*)&hin[(long long)s * F + f];
        acc.x += v.x; acc.y += v.y; acc.z += v.z; acc.w += v.w;
    }
    *(float4*)&agg[(long long)n * F + f] = acc;
}

// -------------------------------------------------- fused GIN layer (WMMA)
// out[n, o] = relu( sum_k (h[n,k] + agg[n,k]*invdeg[n]) * W[o,k] + b[o] )
// Block: 16 nodes x 128 outputs, 8 waves; wave = one 16x16 D-tile,
// v_wmma_f32_16x16x32_f16 over K in steps of 32.

template <int K>
__global__ void __launch_bounds__(256)
gin_layer_kernel(const float* __restrict__ hin,        // [N,K]
                 const float* __restrict__ agg,        // [N,K]
                 const float* __restrict__ invdeg,     // [N]
                 const _Float16* __restrict__ Wh,      // [128,K] row-major
                 const float* __restrict__ bias,       // [128]
                 float* __restrict__ out,              // [N,128]
                 int N) {
    constexpr int LDSK = K + 8;                        // +16B row pad vs bank conflicts
    __shared__ __align__(16) _Float16 zsh[16 * LDSK];

    const int tid   = threadIdx.x;
    const int nbase = blockIdx.x * 16;

    // Phase 1: cooperative z = h + agg*invdeg, f32 -> f16 into LDS.
    for (int idx = tid; idx < 16 * K; idx += 256) {
        int m = idx / K;
        int k = idx - m * K;
        int n = nbase + m;
        float z = 0.0f;
        if (n < N) {
            long long o = (long long)n * K + k;
            z = hin[o] + agg[o] * invdeg[n];
        }
        zsh[m * LDSK + k] = (_Float16)z;
    }
    __syncthreads();

    // Phase 2: per-wave 16x16 WMMA tile.
    const int wave = tid >> 5;
    const int lane = tid & 31;
    const int half = lane >> 4;           // K-half select per ISA fragment layout
    const int row  = lane & 15;           // A-matrix row (node within tile)
    const int col  = wave * 16 + row;     // B/D column (output feature)

    v8f c = {};
#pragma unroll
    for (int kb = 0; kb < K; kb += 32) {
        V16U a, b;
        a.h[0] = *(const v8h*)&zsh[row * LDSK + kb + half * 8];
        a.h[1] = *(const v8h*)&zsh[row * LDSK + kb + 16 + half * 8];
        b.h[0] = *(const v8h*)&Wh[(long long)col * K + kb + half * 8];
        b.h[1] = *(const v8h*)&Wh[(long long)col * K + kb + 16 + half * 8];
        c = __builtin_amdgcn_wmma_f32_16x16x32_f16(
                false, a.v, false, b.v, (short)0, c, false, false);
    }

    // Phase 3: bias + ReLU + store. VGPR r -> M = r + 8*half, N = col.
    const float bc = bias[col];
#pragma unroll
    for (int r = 0; r < 8; ++r) {
        int n = nbase + r + half * 8;
        if (n < N) {
            float v = c[r] + bc;
            out[(long long)n * 128 + col] = v > 0.0f ? v : 0.0f;
        }
    }
}

// ---------------------------------------------------------------- driver

extern "C" void kernel_launch(void* const* d_in, const int* in_sizes, int n_in,
                              void* d_out, int out_size, void* d_ws, size_t ws_size,
                              hipStream_t stream) {
    (void)n_in; (void)out_size; (void)ws_size;
    const float* h   = (const float*)d_in[0];
    const float* W1  = (const float*)d_in[1];
    const float* b1  = (const float*)d_in[2];
    const float* W2  = (const float*)d_in[3];
    const float* b2  = (const float*)d_in[4];
    const int*   src = (const int*)d_in[5];
    const int*   dst = (const int*)d_in[6];

    const int IN = 256, H = 128;
    const int N  = in_sizes[0] / IN;
    const int E  = in_sizes[5];
    float* out = (float*)d_out;

    // Workspace carving (256B-aligned slices).
    char* ws = (char*)d_ws;
    size_t off = 0;
    auto carve = [&](size_t bytes) -> void* {
        void* p = ws + off;
        off = (off + bytes + 255) & ~(size_t)255;
        return p;
    };
    int*      degi   = (int*)carve((size_t)N * sizeof(int));
    float*    invdeg = (float*)carve((size_t)N * sizeof(float));
    int*      rowptr = (int*)carve((size_t)(N + 1) * sizeof(int));
    int*      cursor = (int*)carve((size_t)N * sizeof(int));
    int*      csrsrc = (int*)carve((size_t)E * sizeof(int));
    float*    agg    = (float*)carve((size_t)N * IN * sizeof(float));   // reused for layer 2
    float*    h1     = (float*)carve((size_t)N * H * sizeof(float));
    _Float16* W1h    = (_Float16*)carve((size_t)H * IN * sizeof(_Float16));
    _Float16* W2h    = (_Float16*)carve((size_t)H * H * sizeof(_Float16));

    const int BT = 256;
    auto blk = [&](long long n) -> unsigned { return (unsigned)((n + BT - 1) / BT); };

    // ---- CSR build (once; reused by both layers) + inv_deg
    zero_b128_kernel<<<blk((long long)N / 4), BT, 0, stream>>>((float4*)degi, (long long)N / 4);
    deg_int_kernel<<<blk(E), BT, 0, stream>>>(dst, degi, E);
    invdeg_kernel<<<blk(N), BT, 0, stream>>>(degi, invdeg, N);
    exscan_kernel<<<1, 1024, 0, stream>>>(degi, rowptr, N);
    copy_i32_kernel<<<blk(N), BT, 0, stream>>>(rowptr, cursor, N);
    csr_fill_kernel<<<blk(E), BT, 0, stream>>>(src, dst, cursor, csrsrc, E);

    // f16 weight staging (once, shared by all GEMM blocks through cache)
    f32_to_f16_kernel<<<blk(H * IN), BT, 0, stream>>>(W1, W1h, H * IN);
    f32_to_f16_kernel<<<blk(H * H),  BT, 0, stream>>>(W2, W2h, H * H);

    // ---- layer 1: pull aggregation + fused z/GEMM/bias/relu
    {
        long long total = (long long)N * (IN / 4);
        aggregate_csr_kernel<<<blk(total), BT, 0, stream>>>(h, rowptr, csrsrc, agg, total, IN / 4, IN);
    }
    gin_layer_kernel<256><<<(N + 15) / 16, BT, 0, stream>>>(h, agg, invdeg, W1h, b1, h1, N);

    // ---- layer 2 (reuse agg buffer)
    {
        long long total = (long long)N * (H / 4);
        aggregate_csr_kernel<<<blk(total), BT, 0, stream>>>(h1, rowptr, csrsrc, agg, total, H / 4, H);
    }
    gin_layer_kernel<128><<<(N + 15) / 16, BT, 0, stream>>>(h1, agg, invdeg, W2h, b2, out, N);
}
